// WyzeClassifySTE_3994319585603
// MI455X (gfx1250) — 8-bit matrix operand order repaired and checked by reading, compiled, not run on hardware
//
#include <hip/hip_runtime.h>
#include <hip/hip_bf16.h>
#include <stdint.h>


typedef int v8i __attribute__((ext_vector_type(8)));

#define N_IMG   16
#define C_IN    3
#define H_IN    256
#define W_IN    448
#define C1      64
#define HO      128
#define WO      224
#define C3      128
#define HW      (HO * WO)
#define SPATIAL (N_IMG * HW)
#define MTILES  (SPATIAL / 16)

__device__ __forceinline__ int wrap16i(int v) {
    return (int)((unsigned)(v + 32768) & 0xFFFFu) - 32768;
}

__device__ __forceinline__ float round_rshift_f(float v, int sh) {
    float half = (sh > 0) ? (float)(1 << (sh - 1)) : 0.0f;
    return truncf(ldexpf(v + half, -sh));
}

__global__ void stage1_mulqw(const float* __restrict__ x,
                             const float* __restrict__ w1,
                             const float* __restrict__ b1,
                             const float* __restrict__ s0,
                             const float* __restrict__ s2,
                             int8_t* __restrict__ y)
{
    long idx = (long)blockIdx.x * blockDim.x + threadIdx.x;
    int wo = (int)(idx % WO);
    int ho = (int)((idx / WO) % HO);
    int c  = (int)((idx / ((long)WO * HO)) % C1);
    int n  = (int)(idx / ((long)WO * HO * C1));

    int acc = 0;
    #pragma unroll
    for (int ci = 0; ci < 3; ++ci) {
        #pragma unroll
        for (int kh = 0; kh < 3; ++kh) {
            int hi = ho * 2 + kh;
            #pragma unroll
            for (int kw = 0; kw < 3; ++kw) {
                int wi = wo * 2 + kw;
                int xv = (hi < H_IN && wi < W_IN)
                       ? (int)x[(((long)n * C_IN + ci) * H_IN + hi) * W_IN + wi]
                       : -128;
                int wv = (int)w1[((c * 3 + ci) * 3 + kh) * 3 + kw];
                acc += xv * wv;
            }
        }
    }
    int   a16     = wrap16i(acc);
    float biased  = (float)(a16 + (int)b1[c]);
    float product = biased * s0[c];
    float mq = truncf(ldexpf(product, -31));
    float r  = round_rshift_f(mq, (int)s2[c]);
    float o  = fminf(fmaxf(r, 0.0f), 255.0f) - 128.0f;
    y[idx] = (int8_t)(int)o;
}

__global__ void stage2_mulqh(const int8_t* __restrict__ h1,
                             const float* __restrict__ w2,
                             const float* __restrict__ b2,
                             const float* __restrict__ scl,
                             const float* __restrict__ shr,
                             int8_t* __restrict__ y)
{
    long idx = (long)blockIdx.x * blockDim.x + threadIdx.x;
    int wo = (int)(idx % WO);
    int ho = (int)((idx / WO) % HO);
    int c  = (int)((idx / ((long)WO * HO)) % C1);
    int n  = (int)(idx / ((long)WO * HO * C1));

    int acc = 0;
    #pragma unroll
    for (int kh = 0; kh < 3; ++kh) {
        int hh = ho + kh - 1;
        #pragma unroll
        for (int kw = 0; kw < 3; ++kw) {
            int ww = wo + kw - 1;
            int v = (hh >= 0 && hh < HO && ww >= 0 && ww < WO)
                  ? (int)h1[(((long)n * C1 + c) * HO + hh) * WO + ww]
                  : -128;
            acc += v * (int)w2[c * 9 + kh * 3 + kw];
        }
    }
    acc += (int)b2[c];
    int   a16  = wrap16i(acc);
    float prod = (float)a16 * scl[c];
    float shifted = fminf(fmaxf(prod * 2.0f, -2147483648.0f), 2147483648.0f);
    float mq = truncf(ldexpf(shifted, -16));
    float r  = round_rshift_f(mq, (int)shr[c]);
    float o  = fminf(fmaxf(r, 0.0f), 255.0f) - 128.0f;
    y[((long)n * HW + (long)ho * WO + wo) * 64 + c] = (int8_t)(int)o;
}

__global__ void pack_w3(const float* __restrict__ w3, int8_t* __restrict__ w3p)
{
    int i = blockIdx.x * blockDim.x + threadIdx.x;
    if (i < C3 * C1) w3p[i] = (int8_t)(int)w3[i];
}

__global__ void stage3_mulw_wmma(const int8_t* __restrict__ h2,
                                 const int8_t* __restrict__ w3p,
                                 const float* __restrict__ b3,
                                 const float* __restrict__ scl,
                                 const float* __restrict__ sraw,
                                 const float* __restrict__ srh,
                                 float* __restrict__ out)
{
    const int lane = threadIdx.x & 31;
    const int wave = threadIdx.x >> 5;
    const int tile = blockIdx.x * (blockDim.x >> 5) + wave;
    const int m0   = tile * 16;
    const int half = lane >> 4;
    const int l16  = lane & 15;

    const int2* ap = (const int2*)(h2 + (size_t)(m0 + l16) * 64);
    int2 a01 = ap[0 + half];
    int2 a23 = ap[2 + half];
    int2 a45 = ap[4 + half];
    int2 a67 = ap[6 + half];
    v8i A = { a01.x, a01.y, a23.x, a23.y, a45.x, a45.y, a67.x, a67.y };

    __builtin_prefetch(h2 + (size_t)(m0 + 16 + l16) * 64, 0, 0);

    const int n_img = m0 / HW;
    const int rem   = m0 - n_img * HW;
    const int hrow  = rem / WO;
    const int w0    = rem - hrow * WO + half * 8;

    #pragma unroll
    for (int nt = 0; nt < 8; ++nt) {
        const int occ = nt * 16 + l16;

        const int2* bp = (const int2*)(w3p + (size_t)occ * 64);
        int2 b01 = bp[0 + half];
        int2 b23 = bp[2 + half];
        int2 b45 = bp[4 + half];
        int2 b67 = bp[6 + half];
        v8i B = { b01.x, b01.y, b23.x, b23.y, b45.x, b45.y, b67.x, b67.y };

        v8i Cz = { 0, 0, 0, 0, 0, 0, 0, 0 };
        v8i acc = __builtin_amdgcn_wmma_i32_16x16x64_iu8(
             true, A,  true, B, Cz,
             false,  false);

        const int   bi  = (int)b3[occ];
        const float sc  = scl[occ];
        const int   ssh = (int)sraw[occ];
        const int   rsh = (int)srh[occ];
        float res[8];
        #pragma unroll
        for (int v = 0; v < 8; ++v) {
            int   a16 = wrap16i(acc[v] + bi);
            float p   = (float)a16 * sc;
            float sh  = truncf(ldexpf(p, -ssh));
            float sat = fminf(fmaxf(sh, -32768.0f), 32767.0f);
            float r   = round_rshift_f(sat, rsh);
            res[v]    = fminf(fmaxf(r, 0.0f), 255.0f) - 128.0f;
        }
        float* obase = out + ((((size_t)n_img * C3 + occ) * HO + hrow) * WO + w0);
        *(float4*)(obase + 0) = make_float4(res[0], res[1], res[2], res[3]);
        *(float4*)(obase + 4) = make_float4(res[4], res[5], res[6], res[7]);
    }
}

extern "C" void kernel_launch(void* const* d_in, const int* in_sizes, int n_in,
                              void* d_out, int out_size, void* d_ws, size_t ws_size,
                              hipStream_t stream)
{
    (void)in_sizes; (void)n_in; (void)out_size; (void)ws_size;

    const float* x     = (const float*)d_in[0];
    const float* w1    = (const float*)d_in[1];
    const float* b1    = (const float*)d_in[2];
    const float* s0_1  = (const float*)d_in[3];
    const float* s2_1  = (const float*)d_in[4];
    const float* w2    = (const float*)d_in[5];
    const float* b2    = (const float*)d_in[6];
    const float* scl2  = (const float*)d_in[7];
    const float* sh2   = (const float*)d_in[8];
    const float* w3    = (const float*)d_in[9];
    const float* b3    = (const float*)d_in[10];
    const float* scl3  = (const float*)d_in[11];
    const float* sraw3 = (const float*)d_in[12];
    const float* srh3  = (const float*)d_in[13];
    float* out = (float*)d_out;

    const size_t S = (size_t)SPATIAL * 64;
    int8_t* ws1 = (int8_t*)d_ws;
    int8_t* ws2 = ws1 + S;
    int8_t* w3p = ws2 + S;

    const long elems = (long)N_IMG * C1 * HO * WO;
    const int  blk   = 256;

    pack_w3<<<(C3 * C1 + blk - 1) / blk, blk, 0, stream>>>(w3, w3p);
    stage1_mulqw<<<(int)(elems / blk), blk, 0, stream>>>(x, w1, b1, s0_1, s2_1, ws1);
    stage2_mulqh<<<(int)(elems / blk), blk, 0, stream>>>(ws1, w2, b2, scl2, sh2, ws2);
    stage3_mulw_wmma<<<MTILES / 8, blk, 0, stream>>>(ws2, w3p, b3, scl3, sraw3, srh3, out);
}
